// CrossAttention_7584912245418
// MI455X (gfx1250) — matrix operand order repaired and checked by reading, not compiled, not run on hardware
//
#include <hip/hip_runtime.h>
#include <hip/hip_bf16.h>
#include <stdint.h>

// ---------------------------------------------------------------- types
typedef __bf16 bf16_t;
typedef __attribute__((ext_vector_type(16))) __bf16 v16bf;
typedef __attribute__((ext_vector_type(8)))  __bf16 v8bf;
typedef __attribute__((ext_vector_type(8)))  float  v8f;
typedef __attribute__((ext_vector_type(4)))  float  v4f;
typedef __attribute__((ext_vector_type(4)))  int    v4i_t;

#define WMMA_BF16(A, B, C) \
  __builtin_amdgcn_wmma_f32_16x16x32_bf16(false, (A), false, (B), (short)0, (C), false, false)

// gfx1250 async global->LDS path, feature-guarded so compilation never breaks.
#if defined(__has_builtin)
#if __has_builtin(__builtin_amdgcn_global_load_async_to_lds_b128) && \
    __has_builtin(__builtin_amdgcn_s_wait_asynccnt)
#define HAVE_ASYNC_LDS 1
#endif
#endif

// ---------------------------------------------------------------- problem dims
#define BB   4
#define NN   2048
#define MM   2048
#define QDIM 1024
#define CDIM 1024
#define HH   16
#define DHD  64
#define IDI  1024   // HH*DHD
#define ATT_SCALE 0.125f            // 64^-0.5
#define MAXNEG   (-3.402823466e38f)

// ---------------------------------------------------------------- fragment loaders
// A-matrix 16x32 bf16 (ISA 7.12.2): lane holds row M=lane&15,
// K = {c..c+7, c+16..c+23}, c = (lane>>4)*8.
__device__ __forceinline__ v16bf load_a_bf16(const bf16_t* __restrict__ base, int ld) {
  const int lane = threadIdx.x & 31;
  const bf16_t* p = base + (size_t)(lane & 15) * ld + ((lane >> 4) << 3);
  v8bf lo = *reinterpret_cast<const v8bf*>(p);
  v8bf hi = *reinterpret_cast<const v8bf*>(p + 16);
  v16bf f;
#pragma unroll
  for (int i = 0; i < 8; ++i) { f[i] = lo[i]; f[i + 8] = hi[i]; }
  return f;
}

// Same A layout, source fp32 (converted to bf16 on the fly).
__device__ __forceinline__ v16bf load_a_f32(const float* __restrict__ base, int ld) {
  const int lane = threadIdx.x & 31;
  const float* p = base + (size_t)(lane & 15) * ld + ((lane >> 4) << 3);
  v4f a0 = *reinterpret_cast<const v4f*>(p);
  v4f a1 = *reinterpret_cast<const v4f*>(p + 4);
  v4f a2 = *reinterpret_cast<const v4f*>(p + 16);
  v4f a3 = *reinterpret_cast<const v4f*>(p + 20);
  v16bf f;
#pragma unroll
  for (int i = 0; i < 4; ++i) {
    f[i]      = (bf16_t)a0[i];
    f[i + 4]  = (bf16_t)a1[i];
    f[i + 8]  = (bf16_t)a2[i];
    f[i + 12] = (bf16_t)a3[i];
  }
  return f;
}

// B-matrix 32x16 bf16: lane holds column N=lane&15, K = {c..c+7, c+16..c+23},
// c = (lane>>4)*8. `base` points at row 0 of Bt = B^T stored row-major [N, K].
// Works for both global and LDS source pointers (flat addressing).
__device__ __forceinline__ v16bf load_b_bf16(const bf16_t* base, int ld) {
  const int lane = threadIdx.x & 31;
  const bf16_t* p = base + (size_t)(lane & 15) * ld + ((lane >> 4) << 3);
  v8bf lo = *reinterpret_cast<const v8bf*>(p);
  v8bf hi = *reinterpret_cast<const v8bf*>(p + 16);
  v16bf f;
#pragma unroll
  for (int i = 0; i < 8; ++i) { f[i] = lo[i]; f[i + 8] = hi[i]; }
  return f;
}

// C/D tile store helpers (VGPR j: lanes 0-15 -> M=j, lanes 16-31 -> M=j+8)
__device__ __forceinline__ void store_tile_bf16(bf16_t* __restrict__ C, int Ccols,
                                                int row0, int col0, v8f acc) {
  const int lane = threadIdx.x & 31;
  const int col = col0 + (lane & 15);
  const int rb  = row0 + ((lane >> 4) << 3);
#pragma unroll
  for (int j = 0; j < 8; ++j) C[(size_t)(rb + j) * Ccols + col] = (bf16_t)acc[j];
}

// scatter into vT layout [B, H, DH, M] (attention P*V B-operand)
__device__ __forceinline__ void store_tile_vt(bf16_t* __restrict__ C,
                                              int row0, int col0, v8f acc) {
  const int lane = threadIdx.x & 31;
  const int col = col0 + (lane & 15);
  const int rb  = row0 + ((lane >> 4) << 3);
  const int h = col / DHD, d = col % DHD;
#pragma unroll
  for (int j = 0; j < 8; ++j) {
    int r = rb + j;
    int b = r / MM, jr = r % MM;
    C[((size_t)(b * HH + h) * DHD + d) * MM + jr] = (bf16_t)acc[j];
  }
}

// 16-byte global->LDS stage (async on gfx1250 toolchains that expose it).
// Builtin signature (from compiler diagnostic): param0 = AS1 (global) v4i*,
// param1 = AS3 (shared) v4i*, then imm offset, imm cpol.
__device__ __forceinline__ void stage16(const bf16_t* g, bf16_t* l) {
#if HAVE_ASYNC_LDS
  void* gp = (void*)g;   // shed const; generic
  void* lp = (void*)l;
  __builtin_amdgcn_global_load_async_to_lds_b128(
      (__attribute__((address_space(1))) v4i_t*)gp,
      (__attribute__((address_space(3))) v4i_t*)lp, 0, 0);
#else
  *reinterpret_cast<v8bf*>(l) = *reinterpret_cast<const v8bf*>(g);
#endif
}

__device__ __forceinline__ void stage_wait() {
#if HAVE_ASYNC_LDS
  __builtin_amdgcn_s_wait_asynccnt(0);
#endif
}

// ---------------------------------------------------------------- kernel 0: W fp32 -> Wt bf16
// dst[c*Krows + k] = (bf16) src[k*Ccols + c]
__global__ __launch_bounds__(256) void transpose_to_bf16(
    const float* __restrict__ src, bf16_t* __restrict__ dst, int Krows, int Ccols) {
  int idx = blockIdx.x * 256 + threadIdx.x;
  int c = idx % Ccols;
  int k = idx / Ccols;
  dst[(size_t)c * Krows + k] = (bf16_t)src[(size_t)k * Ccols + c];
}

// ---------------------------------------------------------------- kernel 1: NT GEMM, A fp32
// 32x32 tile per wave (2x2 WMMA): 2x fragment reuse.
__global__ __launch_bounds__(256) void gemm_f32a_bf16o(
    const float* __restrict__ A, const bf16_t* __restrict__ Bt,
    bf16_t* __restrict__ C, int Ccols, int K, int vt) {
  const int wave   = blockIdx.x * 8 + (threadIdx.x >> 5);
  const int tilesC = Ccols >> 5;
  const int tr = wave / tilesC, tc = wave % tilesC;

  const float*  A0 = A  + (size_t)tr * 32 * K;
  const float*  A1 = A0 + (size_t)16 * K;
  const bf16_t* B0 = Bt + (size_t)tc * 32 * K;
  const bf16_t* B1 = B0 + (size_t)16 * K;

  v8f c00 = {}, c01 = {}, c10 = {}, c11 = {};
  for (int kb = 0; kb < K; kb += 32) {
    v16bf a0 = load_a_f32(A0 + kb, K);
    v16bf a1 = load_a_f32(A1 + kb, K);
    v16bf b0 = load_b_bf16(B0 + kb, K);
    v16bf b1 = load_b_bf16(B1 + kb, K);
    c00 = WMMA_BF16(a0, b0, c00);
    c01 = WMMA_BF16(a0, b1, c01);
    c10 = WMMA_BF16(a1, b0, c10);
    c11 = WMMA_BF16(a1, b1, c11);
  }

  const int r0 = tr * 32, c0 = tc * 32;
  if (!vt) {
    store_tile_bf16(C, Ccols, r0,      c0,      c00);
    store_tile_bf16(C, Ccols, r0,      c0 + 16, c01);
    store_tile_bf16(C, Ccols, r0 + 16, c0,      c10);
    store_tile_bf16(C, Ccols, r0 + 16, c0 + 16, c11);
  } else {
    store_tile_vt(C, r0,      c0,      c00);
    store_tile_vt(C, r0,      c0 + 16, c01);
    store_tile_vt(C, r0 + 16, c0,      c10);
    store_tile_vt(C, r0 + 16, c0 + 16, c11);
  }
}

// ---------------------------------------------------------------- kernel 2: flash attention
// block = 8 waves sharing (b,h); K/V chunks staged cooperatively in LDS
// (async global->LDS when available), 8x reuse across the block's waves.
__global__ __launch_bounds__(256) void attn_flash(
    const bf16_t* __restrict__ qb, const bf16_t* __restrict__ kb,
    const bf16_t* __restrict__ vT, const uint8_t* __restrict__ mask,
    bf16_t* __restrict__ attn) {
  __shared__ __align__(32) bf16_t kTile[32 * 64];   // rows j-local, cols d   (4 KB)
  __shared__ __align__(32) bf16_t vTile[64 * 32];   // rows d,       cols j   (4 KB)
  __shared__ __align__(16) bf16_t plds[8][16 * 32]; // per-wave P staging     (8 KB)

  const int tid  = threadIdx.x;
  const int wid  = tid >> 5;
  const int lane = tid & 31;
  const int lm   = lane & 15;
  const int lko  = (lane >> 4) << 3;

  const int qblocks = (NN / 16) / 8;          // 16 blocks per (b,h)
  const int qt = (blockIdx.x % qblocks) * 8 + wid;
  const int h  = (blockIdx.x / qblocks) % HH;
  const int b  = blockIdx.x / (qblocks * HH);

  const bf16_t* qbase = qb + ((size_t)(b * NN + qt * 16)) * IDI + h * DHD;
  const v16bf qa0 = load_a_bf16(qbase,      IDI);   // d = 0..31
  const v16bf qa1 = load_a_bf16(qbase + 32, IDI);   // d = 32..63

  const bf16_t* kbh = kb + (size_t)b * MM * IDI + h * DHD;
  const bf16_t* vbh = vT + ((size_t)(b * HH + h)) * DHD * MM;
  const uint8_t* mb = mask + (size_t)b * MM;

  // cooperative stage indices: K = 32 rows x 128B, V = 64 rows x 64B
  const int krow = tid >> 3, kseg = tid & 7;
  const int vrow = tid >> 2, vseg = tid & 3;

  v8f o0 = {}, o1 = {}, o2 = {}, o3 = {};
  float mrow[8], lrow[8];
#pragma unroll
  for (int j = 0; j < 8; ++j) { mrow[j] = MAXNEG; lrow[j] = 0.f; }

  bf16_t* pl = &plds[wid][0];

  for (int j0 = 0; j0 < MM; j0 += 32) {
    // ---- stage K/V chunk into LDS (once per block, shared by 8 waves) ----
    stage16(kbh + (size_t)(j0 + krow) * IDI + kseg * 8, &kTile[krow * 64 + kseg * 8]);
    stage16(vbh + (size_t)vrow * MM + j0 + vseg * 8,    &vTile[vrow * 32 + vseg * 8]);
    stage_wait();
    __syncthreads();

    // ---- S = (Q K^T) * scale, masked -------------------------------------
    v8f s0 = {}, s1 = {};
    s0 = WMMA_BF16(qa0, load_b_bf16(kTile,                64), s0);
    s0 = WMMA_BF16(qa1, load_b_bf16(kTile + 32,           64), s0);
    s1 = WMMA_BF16(qa0, load_b_bf16(kTile + 16 * 64,      64), s1);
    s1 = WMMA_BF16(qa1, load_b_bf16(kTile + 16 * 64 + 32, 64), s1);

    const bool ok0 = mb[j0 + lm] != 0;
    const bool ok1 = mb[j0 + 16 + lm] != 0;
#pragma unroll
    for (int j = 0; j < 8; ++j) {
      float a = s0[j] * ATT_SCALE; s0[j] = ok0 ? a : MAXNEG;
      float c = s1[j] * ATT_SCALE; s1[j] = ok1 ? c : MAXNEG;
    }

    // ---- online softmax (row r = j + (lane>>4)*8; 16-lane groups) --------
#pragma unroll
    for (int j = 0; j < 8; ++j) {
      float v = fmaxf(s0[j], s1[j]);
#pragma unroll
      for (int off = 1; off < 16; off <<= 1) v = fmaxf(v, __shfl_xor(v, off, 32));
      float mnew = fmaxf(mrow[j], v);
      float corr = __expf(mrow[j] - mnew);
      float p0 = __expf(s0[j] - mnew);
      float p1 = __expf(s1[j] - mnew);
      float ps = p0 + p1;
#pragma unroll
      for (int off = 1; off < 16; off <<= 1) ps += __shfl_xor(ps, off, 32);
      lrow[j] = lrow[j] * corr + ps;
      mrow[j] = mnew;
      o0[j] *= corr; o1[j] *= corr; o2[j] *= corr; o3[j] *= corr;
      const int row = j + ((lane >> 4) << 3);
      pl[row * 32 + lm]      = (bf16_t)p0;
      pl[row * 32 + 16 + lm] = (bf16_t)p1;
    }

    // ---- P (C/D layout) -> A layout via per-wave LDS ---------------------
    v16bf pa;
#pragma unroll
    for (int i = 0; i < 8; ++i) {
      pa[i]     = pl[lm * 32 + lko + i];
      pa[i + 8] = pl[lm * 32 + lko + 16 + i];
    }

    // ---- O += P * V ------------------------------------------------------
    o0 = WMMA_BF16(pa, load_b_bf16(vTile + 0 * 16 * 32, 32), o0);
    o1 = WMMA_BF16(pa, load_b_bf16(vTile + 1 * 16 * 32, 32), o1);
    o2 = WMMA_BF16(pa, load_b_bf16(vTile + 2 * 16 * 32, 32), o2);
    o3 = WMMA_BF16(pa, load_b_bf16(vTile + 3 * 16 * 32, 32), o3);

    __syncthreads();   // protect kTile/vTile before next stage
  }

  // ---- finalize and store attention tile [16 x 64] bf16 ------------------
  bf16_t* outp = attn + ((size_t)(b * NN + qt * 16)) * IDI + h * DHD;
#pragma unroll
  for (int j = 0; j < 8; ++j) {
    const float inv = 1.f / lrow[j];
    const int row = j + ((lane >> 4) << 3);
    outp[(size_t)row * IDI      + lm] = (bf16_t)(o0[j] * inv);
    outp[(size_t)row * IDI + 16 + lm] = (bf16_t)(o1[j] * inv);
    outp[(size_t)row * IDI + 32 + lm] = (bf16_t)(o2[j] * inv);
    outp[(size_t)row * IDI + 48 + lm] = (bf16_t)(o3[j] * inv);
  }
}

// ---------------------------------------------------------------- kernel 3: NT GEMM, A bf16 -> fp32 + bias
__global__ __launch_bounds__(256) void gemm_bf16a_f32o_bias(
    const bf16_t* __restrict__ A, const bf16_t* __restrict__ Bt,
    const float* __restrict__ bias, float* __restrict__ C, int Ccols, int K) {
  const int wave   = blockIdx.x * 8 + (threadIdx.x >> 5);
  const int tilesC = Ccols >> 5;
  const int tr = wave / tilesC, tc = wave % tilesC;
  const int lane = threadIdx.x & 31;

  const bf16_t* A0 = A  + (size_t)tr * 32 * K;
  const bf16_t* A1 = A0 + (size_t)16 * K;
  const bf16_t* B0 = Bt + (size_t)tc * 32 * K;
  const bf16_t* B1 = B0 + (size_t)16 * K;

  v8f c00 = {}, c01 = {}, c10 = {}, c11 = {};
  for (int kb = 0; kb < K; kb += 32) {
    v16bf a0 = load_a_bf16(A0 + kb, K);
    v16bf a1 = load_a_bf16(A1 + kb, K);
    v16bf b0 = load_b_bf16(B0 + kb, K);
    v16bf b1 = load_b_bf16(B1 + kb, K);
    c00 = WMMA_BF16(a0, b0, c00);
    c01 = WMMA_BF16(a0, b1, c01);
    c10 = WMMA_BF16(a1, b0, c10);
    c11 = WMMA_BF16(a1, b1, c11);
  }

  const int colL = lane & 15;
  const int rsel = (lane >> 4) << 3;
  const int r0 = tr * 32, c0 = tc * 32;
#pragma unroll
  for (int t = 0; t < 4; ++t) {
    const v8f acc = (t == 0) ? c00 : (t == 1) ? c01 : (t == 2) ? c10 : c11;
    const int col = c0 + ((t & 1) << 4) + colL;
    const int rb  = r0 + ((t >> 1) << 4) + rsel;
    const float bo = bias[col];
#pragma unroll
    for (int j = 0; j < 8; ++j)
      C[(size_t)(rb + j) * Ccols + col] = acc[j] + bo;
  }
}

// ---------------------------------------------------------------- host launcher
extern "C" void kernel_launch(void* const* d_in, const int* in_sizes, int n_in,
                              void* d_out, int out_size, void* d_ws, size_t ws_size,
                              hipStream_t stream) {
  (void)in_sizes; (void)n_in; (void)out_size; (void)ws_size;

  const float*   x    = (const float*)d_in[0];
  const float*   ctx  = (const float*)d_in[1];
  const uint8_t* mask = (const uint8_t*)d_in[2];   // numpy bool = 1 byte
  const float*   Wq   = (const float*)d_in[3];
  const float*   Wk   = (const float*)d_in[4];
  const float*   Wv   = (const float*)d_in[5];
  const float*   Wo   = (const float*)d_in[6];
  const float*   bo   = (const float*)d_in[7];
  float*         out  = (float*)d_out;

  // workspace carve (bf16 elements): 4x 1M weights + 4x 8M activations = 72 MB
  bf16_t* p = (bf16_t*)d_ws;
  bf16_t* Wqt = p; p += (size_t)IDI * QDIM;
  bf16_t* Wkt = p; p += (size_t)IDI * CDIM;
  bf16_t* Wvt = p; p += (size_t)IDI * CDIM;
  bf16_t* Wot = p; p += (size_t)QDIM * IDI;
  bf16_t* qb  = p; p += (size_t)BB * NN * IDI;
  bf16_t* kbf = p; p += (size_t)BB * MM * IDI;
  bf16_t* vTb = p; p += (size_t)BB * MM * IDI;
  bf16_t* att = p; p += (size_t)BB * NN * IDI;

  const dim3 blk(256);

  // 0) weights -> transposed bf16
  const int tgrid = (QDIM * IDI) / 256;   // 4096
  transpose_to_bf16<<<tgrid, blk, 0, stream>>>(Wq, Wqt, QDIM, IDI);
  transpose_to_bf16<<<tgrid, blk, 0, stream>>>(Wk, Wkt, CDIM, IDI);
  transpose_to_bf16<<<tgrid, blk, 0, stream>>>(Wv, Wvt, CDIM, IDI);
  transpose_to_bf16<<<tgrid, blk, 0, stream>>>(Wo, Wot, IDI, QDIM);

  // 1) projections: 32x32 tiles -> (8192/32)*(1024/32)=8192 tiles -> 1024 blocks
  const int ggrid = ((size_t)BB * NN * IDI) / (32 * 32 * 8);   // 1024
  gemm_f32a_bf16o<<<ggrid, blk, 0, stream>>>(x,   Wqt, qb,  IDI, QDIM, 0);
  gemm_f32a_bf16o<<<ggrid, blk, 0, stream>>>(ctx, Wkt, kbf, IDI, CDIM, 0);
  gemm_f32a_bf16o<<<ggrid, blk, 0, stream>>>(ctx, Wvt, vTb, IDI, CDIM, 1);

  // 2) flash attention: B*H*16 = 1024 blocks of 8 waves (shared K/V staging)
  const int agrid = BB * HH * ((NN / 16) / 8);                 // 1024
  attn_flash<<<agrid, blk, 0, stream>>>(qb, kbf, vTb, mask, att);

  // 3) output projection + bias -> fp32 d_out
  gemm_bf16a_f32o_bias<<<ggrid, blk, 0, stream>>>(att, Wot, bo, out, QDIM, IDI);
}
